// DFlashAttention_11622181503681
// MI455X (gfx1250) — compile-verified, matrix operand order checked by reading, not run on hardware
//
#include <hip/hip_runtime.h>
#include <cstdint>
#include <cstddef>

// ---------------------------------------------------------------------------
// Types for CDNA5 WMMA (wave32): v_wmma_f32_16x16x32_bf16
// ---------------------------------------------------------------------------
typedef __attribute__((ext_vector_type(16))) __bf16 v16bf;
typedef __attribute__((ext_vector_type(8)))  __bf16 bf16x8;
typedef __attribute__((ext_vector_type(8)))  float  v8f;
typedef __attribute__((ext_vector_type(4)))  unsigned int u32x4;
typedef __attribute__((ext_vector_type(8)))  int          i32x8;
typedef __attribute__((ext_vector_type(4)))  int          i32x4;

#define S_LEN 2048
#define HID   2048
#define NH    16
#define NKV   4
#define HD    128

// Tensor Data Mover availability (ROCm7.2 clang-22: 5 args; clang-23: 6 args)
#if defined(__has_builtin)
#if __has_builtin(__builtin_amdgcn_tensor_load_to_lds) && \
    __has_builtin(__builtin_amdgcn_s_wait_tensorcnt)
#define HAVE_TDM 1
#endif
#endif
#ifndef HAVE_TDM
#define HAVE_TDM 0
#endif

__device__ __forceinline__ unsigned short f2bf(float x) {
  unsigned int u = __float_as_uint(x);
  u += 0x7FFFu + ((u >> 16) & 1u);            // round-to-nearest-even
  return (unsigned short)(u >> 16);
}

__device__ __forceinline__ v16bf mk16(bf16x8 lo, bf16x8 hi) {
  return __builtin_shufflevector(lo, hi, 0,1,2,3,4,5,6,7,8,9,10,11,12,13,14,15);
}

// Load one A/B fragment (16 bf16 per lane): caller passes pointer to this
// lane's k-range start (row*stride + 8*half). ISA 16-bit layout:
// elems 0..7 -> k = 8*half + e ; elems 8..15 -> k = 16 + 8*half + (e-8).
__device__ __forceinline__ v16bf ldfrag(const unsigned short* p) {
  bf16x8 lo = *reinterpret_cast<const bf16x8*>(p);
  bf16x8 hi = *reinterpret_cast<const bf16x8*>(p + 16);
  return mk16(lo, hi);
}

__device__ __forceinline__ float red16max(float v) {
#pragma unroll
  for (int m = 1; m < 16; m <<= 1) v = fmaxf(v, __shfl_xor(v, m, 32));
  return v;
}
__device__ __forceinline__ float red16sum(float v) {
#pragma unroll
  for (int m = 1; m < 16; m <<= 1) v += __shfl_xor(v, m, 32);
  return v;
}

#if HAVE_TDM
// Issue a 2D TDM tile load: D# group0 built from LDS offset + global address,
// group1 passed fully-constant from the call site (ISA §8 bitfields).
__device__ __forceinline__ void tdm_load_2d(unsigned int lds_off, const void* g,
                                            i32x8 g1) {
  unsigned long long ga = (unsigned long long)(size_t)g;
  u32x4 g0;
  g0[0] = 1u;                                   // count=1 valid descriptor
  g0[1] = lds_off;                              // lds_addr (bytes)
  g0[2] = (unsigned int)ga;                     // global_addr[31:0]
  g0[3] = ((unsigned int)(ga >> 32) & 0x01FFFFFFu) | (2u << 30); // addr[56:32] | type=2
  i32x4 z4 = {0, 0, 0, 0};
#if __clang_major__ >= 23
  i32x8 z8 = {0, 0, 0, 0, 0, 0, 0, 0};
  __builtin_amdgcn_tensor_load_to_lds(g0, g1, z4, z4, z8, 0);
#else
  __builtin_amdgcn_tensor_load_to_lds(g0, g1, z4, z4, 0);
#endif
}
#endif

// ---------------------------------------------------------------------------
// GEMM: C[M][N](f32) = A[M][KD](f32->bf16) @ B[KD][N](f32->bf16)
// Block tile 128x128, 8 waves, wave tile 32(M) x 64(N), K-step 32.
// N, KD compile-time so the epilogue uses immediate store offsets.
// ---------------------------------------------------------------------------
template <int N, int KD>
__global__ __launch_bounds__(256) void gemm_bf16(const float* __restrict__ A,
                                                 const float* __restrict__ Bm,
                                                 float* __restrict__ C) {
  const int tid  = threadIdx.x;
  const int w    = tid >> 5;
  const int l    = tid & 31;
  const int half = l >> 4;
  const int lm   = l & 15;
  const int m0   = blockIdx.y * 128;
  const int n0   = blockIdx.x * 128;
  const int wm   = (w & 3) * 32;
  const int wn   = (w >> 2) * 64;

  __shared__ alignas(16) unsigned short As[128 * 32];   // [m][k] bf16
  __shared__ alignas(16) unsigned short Bs[128 * 32];   // [n][k] bf16 (B transposed)

  v8f acc[2][4];
#pragma unroll
  for (int mi = 0; mi < 2; mi++)
#pragma unroll
    for (int ni = 0; ni < 4; ni++) acc[mi][ni] = (v8f){};

  for (int k0 = 0; k0 < KD; k0 += 32) {
    // Stage A tile: 128 rows x 32 cols, float4/thread x4, cvt->bf16
#pragma unroll
    for (int it = 0; it < 4; it++) {
      int c   = it * 256 + tid;
      int row = c >> 3;
      int col = (c & 7) * 4;
      const float* src = A + (size_t)(m0 + row) * KD + k0 + col;
      const float4 f = *reinterpret_cast<const float4*>(src);
      if (k0 + 32 < KD) __builtin_prefetch(src + 32, 0, 3);  // global_prefetch_b8
      unsigned short* d = &As[row * 32 + col];
      d[0] = f2bf(f.x); d[1] = f2bf(f.y); d[2] = f2bf(f.z); d[3] = f2bf(f.w);
    }
    // Stage B tile transposed: Bs[n][k] = B[k0+k][n0+n]
#pragma unroll
    for (int it = 0; it < 4; it++) {
      int c  = it * 256 + tid;
      int k  = c >> 5;
      int n  = (c & 31) * 4;
      const float* src = Bm + (size_t)(k0 + k) * N + n0 + n;
      const float4 f = *reinterpret_cast<const float4*>(src);
      if (k0 + 32 < KD) __builtin_prefetch(src + (size_t)32 * N, 0, 3);
      Bs[(n + 0) * 32 + k] = f2bf(f.x);
      Bs[(n + 1) * 32 + k] = f2bf(f.y);
      Bs[(n + 2) * 32 + k] = f2bf(f.z);
      Bs[(n + 3) * 32 + k] = f2bf(f.w);
    }
    __syncthreads();

    v16bf af[2], bf[4];
#pragma unroll
    for (int mi = 0; mi < 2; mi++)
      af[mi] = ldfrag(&As[(wm + mi * 16 + lm) * 32 + 8 * half]);
#pragma unroll
    for (int ni = 0; ni < 4; ni++)
      bf[ni] = ldfrag(&Bs[(wn + ni * 16 + lm) * 32 + 8 * half]);

#pragma unroll
    for (int mi = 0; mi < 2; mi++)
#pragma unroll
      for (int ni = 0; ni < 4; ni++)
        acc[mi][ni] = __builtin_amdgcn_wmma_f32_16x16x32_bf16(
            false, af[mi], false, bf[ni], (short)0, acc[mi][ni], false, false);
    __syncthreads();
  }

  // C layout: lane holds col n = lm, VGPR r holds row m = r + 8*half.
  // One base pointer; all offsets compile-time -> immediate-offset stores.
  float* Cp = C + (size_t)(m0 + wm + 8 * half) * N + (n0 + wn + lm);
#pragma unroll
  for (int mi = 0; mi < 2; mi++)
#pragma unroll
    for (int ni = 0; ni < 4; ni++)
#pragma unroll
      for (int r = 0; r < 8; r++)
        Cp[(size_t)(mi * 16 + r) * N + ni * 16] = acc[mi][ni][r];
}

// ---------------------------------------------------------------------------
// pack_qkv: per (b,s,slot) block of 128 threads.
// slot 0..15 = q head (norm+rope -> Q bf16 [B][H][S][D])
// slot 16..19 = k head (norm+rope -> K bf16 [B][KV][S][D])
// slot 20..23 = v head (cast -> Vt bf16 [B][KV][D][S], pre-transposed)
// ---------------------------------------------------------------------------
__global__ __launch_bounds__(128) void pack_qkv(const float* __restrict__ qkv,
                                                const int* __restrict__ pos,
                                                const float* __restrict__ qw,
                                                const float* __restrict__ kw,
                                                unsigned short* __restrict__ Q,
                                                unsigned short* __restrict__ K,
                                                unsigned short* __restrict__ Vt) {
  const int slot = blockIdx.x % 24;
  const int bs   = blockIdx.x / 24;
  const int b    = bs / S_LEN;
  const int s    = bs % S_LEN;
  const int d    = threadIdx.x;

  const float* row = qkv + (size_t)bs * 3072;
  float val;
  if (slot < 16)       val = row[slot * HD + d];
  else if (slot < 20)  val = row[2048 + (slot - 16) * HD + d];
  else                 val = row[2560 + (slot - 20) * HD + d];

  if (slot >= 20) {  // V: plain cast, transposed store (uniform branch)
    Vt[((size_t)(b * NKV + (slot - 20)) * HD + d) * S_LEN + s] = f2bf(val);
    return;
  }

  // RMSNorm over 128 dims (4 waves)
  float ss = val * val;
#pragma unroll
  for (int m = 1; m < 32; m <<= 1) ss += __shfl_xor(ss, m, 32);
  __shared__ float red[4];
  if ((threadIdx.x & 31) == 0) red[threadIdx.x >> 5] = ss;
  __syncthreads();
  float total = red[0] + red[1] + red[2] + red[3];
  float wgt   = (slot < 16) ? qw[d] : kw[d];
  float nv    = val * rsqrtf(total * (1.0f / HD) + 1e-6f) * wgt;

  // RoPE (half-rotation): needs partner x[d^64]
  __shared__ float xs[128];
  xs[d] = nv;
  __syncthreads();
  float partner = xs[d ^ 64];
  int   i    = d & 63;
  float p    = (float)pos[bs];
  float invf = __expf(-(float)i * (13.815510558f / 64.0f));  // 1e6^(-2i/128)
  float ang  = p * invf;
  float cv   = __cosf(ang), sv = __sinf(ang);
  float out  = (d < 64) ? nv * cv - partner * sv : nv * cv + partner * sv;

  if (slot < 16)
    Q[((size_t)(b * NH + slot) * S_LEN + s) * HD + d] = f2bf(out);
  else
    K[((size_t)(b * NKV + (slot - 16)) * S_LEN + s) * HD + d] = f2bf(out);
}

// ---------------------------------------------------------------------------
// Flash attention. 4 waves/block share (b,h): wave w owns q rows s0..s0+15.
// K/V tiles for each 32-key step are staged into LDS by the Tensor Data
// Mover (double-buffered, s_wait_tensorcnt + barriers); TDM LDS padding
// breaks the 64-byte-multiple row stride to avoid bank conflicts.
// Q bf16 [B][H][S][D], K bf16 [B][KV][S][D], Vt bf16 [B][KV][D][S].
// ---------------------------------------------------------------------------
#define KROW 136   // 128 bf16 + 8 pad (TDM pad: 4 dwords per 64-dword row)
#define VROW 40    // 32 bf16 + 8 pad  (TDM pad: 4 dwords per 16-dword row)

__global__ __launch_bounds__(128) void attn_kernel(const unsigned short* __restrict__ Q,
                                                   const unsigned short* __restrict__ K,
                                                   const unsigned short* __restrict__ Vt,
                                                   float* __restrict__ O) {
  const int w    = threadIdx.x >> 5;
  const int l    = threadIdx.x & 31;
  const int half = l >> 4;
  const int lm   = l & 15;

  const int t64 = blockIdx.x & 31;            // S/64 tiles
  const int h   = (blockIdx.x >> 5) & 15;
  const int b   = blockIdx.x >> 9;
  const int s0  = t64 * 64 + w * 16;
  const int kv  = h >> 2;                     // GQA: 4 q heads per kv head

  const unsigned short* Qb = Q + ((size_t)(b * NH + h) * S_LEN + s0) * HD;
  const unsigned short* Kb = K + (size_t)(b * NKV + kv) * S_LEN * HD;
  const unsigned short* Vb = Vt + (size_t)(b * NKV + kv) * HD * S_LEN;

  __shared__ alignas(16) unsigned short Pt[4][16 * 32];  // per-wave P staging
  unsigned short* Pw = Pt[w];

#if HAVE_TDM
  __shared__ alignas(16) unsigned short Ks[2][32 * KROW];   // [key][d] padded
  __shared__ alignas(16) unsigned short Vs[2][128 * VROW];  // [d][key] padded
  // D# group1 constants (ISA §8.4): 2D tile, data_size=2B, pad enabled.
  // K tile: tensor_dim0=128, tensor_dim1=2048, tile=128x32, stride0=128,
  //         pad_interval=64dw(code5), pad_amount=4dw(code3)
  const i32x8 kg1 = {0x07510000, 128 << 16, 2048 << 16, 128 << 16, 32, 128, 0, 0};
  // V tile: tensor_dim0=2048, tensor_dim1=128, tile=32x128, stride0=2048,
  //         pad_interval=16dw(code3), pad_amount=4dw(code3)
  const i32x8 vg1 = {0x06D10000, 2048 << 16, 128 << 16, 32 << 16, 128, 2048, 0, 0};
#endif

  // Q fragments (16 rows x 128 dims = 4 K-chunks), kept in registers
  v16bf qf[4];
#pragma unroll
  for (int kc = 0; kc < 4; kc++)
    qf[kc] = ldfrag(Qb + lm * HD + kc * 32 + 8 * half);

  v8f acc[8];
#pragma unroll
  for (int t = 0; t < 8; t++) acc[t] = (v8f){};
  float rm[8], rl[8];
#pragma unroll
  for (int r = 0; r < 8; r++) { rm[r] = -INFINITY; rl[r] = 0.0f; }

  const float scale = 0.08838834764831845f;   // 1/sqrt(128)
  const int   NT    = S_LEN / 32;

#if HAVE_TDM
  if (w == 0) {  // prime buffer 0 (TDM ignores EXEC; one issue per wave)
    tdm_load_2d((unsigned int)(size_t)&Ks[0][0], Kb, kg1);
    tdm_load_2d((unsigned int)(size_t)&Vs[0][0], Vb, vg1);
  }
#endif

  for (int jt = 0; jt < NT; jt++) {
    const int j   = jt * 32;
    const int buf = jt & 1;
#if HAVE_TDM
    if (w == 0 && jt + 1 < NT) {  // issue next tile into other buffer
      tdm_load_2d((unsigned int)(size_t)&Ks[buf ^ 1][0],
                  Kb + (size_t)(j + 32) * HD, kg1);
      tdm_load_2d((unsigned int)(size_t)&Vs[buf ^ 1][0], Vb + j + 32, vg1);
    }
    // TDM completes in order: <=2 outstanding means current tile is ready.
    if (jt + 1 < NT) __builtin_amdgcn_s_wait_tensorcnt(2);
    else             __builtin_amdgcn_s_wait_tensorcnt(0);
    __syncthreads();
#endif

    // scores: 16 x 32 tile as two 16x16 WMMA C-tiles
    v8f sc[2];
#pragma unroll
    for (int sub = 0; sub < 2; sub++) {
      v8f c = (v8f){};
#pragma unroll
      for (int kc = 0; kc < 4; kc++) {
#if HAVE_TDM
        v16bf kf = ldfrag(&Ks[buf][(sub * 16 + lm) * KROW + kc * 32 + 8 * half]);
#else
        v16bf kf = ldfrag(Kb + (size_t)(j + sub * 16 + lm) * HD + kc * 32 + 8 * half);
#endif
        c = __builtin_amdgcn_wmma_f32_16x16x32_bf16(
            false, qf[kc], false, kf, (short)0, c, false, false);
      }
      sc[sub] = c;
    }

    // online softmax per row (row m = r + 8*half, cols across 16 lanes of half)
#pragma unroll
    for (int r = 0; r < 8; r++) {
      float s0v = sc[0][r] * scale;
      float s1v = sc[1][r] * scale;
      float mx    = red16max(fmaxf(s0v, s1v));
      float mnew  = fmaxf(rm[r], mx);
      float alpha = __expf(rm[r] - mnew);
      float p0    = __expf(s0v - mnew);
      float p1    = __expf(s1v - mnew);
      float rs    = red16sum(p0 + p1);
      rl[r] = rl[r] * alpha + rs;
      rm[r] = mnew;
      int m = r + 8 * half;
      Pw[m * 32 + lm]      = f2bf(p0);
      Pw[m * 32 + 16 + lm] = f2bf(p1);
#pragma unroll
      for (int t = 0; t < 8; t++) acc[t][r] *= alpha;
    }

    // P (C layout -> A layout) via wave-private LDS; DS ops in-order per wave
    v16bf pf = ldfrag(Pw + lm * 32 + 8 * half);

    // acc += P @ V  (B operand: rows = output dims, cols = keys)
#pragma unroll
    for (int t = 0; t < 8; t++) {
#if HAVE_TDM
      v16bf vf = ldfrag(&Vs[buf][(t * 16 + lm) * VROW + 8 * half]);
#else
      v16bf vf = ldfrag(Vb + (size_t)(t * 16 + lm) * S_LEN + j + 8 * half);
#endif
      acc[t] = __builtin_amdgcn_wmma_f32_16x16x32_bf16(
          false, pf, false, vf, (short)0, acc[t], false, false);
    }
#if HAVE_TDM
    __syncthreads();  // all waves done with buf before it is refilled
#endif
  }

  // epilogue: divide by softmax denominator, store f32 [B][S][H*D]
#pragma unroll
  for (int r = 0; r < 8; r++) {
    float inv = 1.0f / rl[r];
    int   m   = r + 8 * half;
    size_t rowoff = ((size_t)(b * S_LEN) + s0 + m) * (NH * HD) + h * HD;
#pragma unroll
    for (int t = 0; t < 8; t++)
      O[rowoff + t * 16 + lm] = acc[t][r] * inv;
  }
}

// ---------------------------------------------------------------------------
// Launch: qkv GEMM -> norm/rope/pack -> flash attention -> out GEMM
// Workspace: [0, 48MB) qkv f32 (reused as attn f32), then Q/K/Vt bf16.
// ---------------------------------------------------------------------------
extern "C" void kernel_launch(void* const* d_in, const int* in_sizes, int n_in,
                              void* d_out, int out_size, void* d_ws, size_t ws_size,
                              hipStream_t stream) {
  const float* hidden = (const float*)d_in[0];
  const int*   pos    = (const int*)d_in[1];
  const float* Wqkv   = (const float*)d_in[2];
  const float* Wo     = (const float*)d_in[3];
  const float* qw     = (const float*)d_in[4];
  const float* kw     = (const float*)d_in[5];
  float*       out    = (float*)d_out;

  char* ws = (char*)d_ws;
  const size_t QKV_BYTES = (size_t)4096 * 3072 * 4;          // 48 MB
  const size_t Q_BYTES   = (size_t)2 * NH * S_LEN * HD * 2;  // 16 MB
  const size_t K_BYTES   = (size_t)2 * NKV * S_LEN * HD * 2; //  4 MB
  float*          qkv = (float*)ws;
  unsigned short* Qb  = (unsigned short*)(ws + QKV_BYTES);
  unsigned short* Kb  = (unsigned short*)(ws + QKV_BYTES + Q_BYTES);
  unsigned short* Vt  = (unsigned short*)(ws + QKV_BYTES + Q_BYTES + K_BYTES);
  float*          attn = qkv;  // qkv region dead after pack_qkv

  // 1) qkv = hidden @ Wqkv   (M=4096, N=3072, K=2048)
  gemm_bf16<3072, 2048><<<dim3(3072 / 128, 4096 / 128), 256, 0, stream>>>(
      hidden, Wqkv, qkv);

  // 2) RMSNorm + RoPE + pack to bf16 (V pre-transposed)
  pack_qkv<<<dim3(2 * S_LEN * 24), 128, 0, stream>>>(qkv, pos, qw, kw, Qb, Kb, Vt);

  // 3) flash attention (grid = B * H * S/64)
  attn_kernel<<<dim3(2 * NH * (S_LEN / 64)), 128, 0, stream>>>(Qb, Kb, Vt, attn);

  // 4) out = attn @ Wo       (M=4096, N=2048, K=2048)
  gemm_bf16<2048, 2048><<<dim3(2048 / 128, 4096 / 128), 256, 0, stream>>>(
      attn, Wo, out);
}
